// TantraAttention_45234595561841
// MI455X (gfx1250) — matrix operand order repaired and checked by reading, not compiled, not run on hardware
//
#include <hip/hip_runtime.h>
#include <hip/hip_bf16.h>
#include <stdint.h>

#define D_MODEL 1024
#define N_HEADS 16
#define HEAD_DIM 64
#define BATCH 4
#define SEQ 2048
#define MROWS (BATCH * SEQ) /* 8192 */

typedef __attribute__((ext_vector_type(16))) __bf16          v16bf;
typedef __attribute__((ext_vector_type(8)))  float           v8f;
typedef __attribute__((ext_vector_type(8)))  unsigned int    v8u;
typedef __attribute__((ext_vector_type(8)))  unsigned short  u16x8;

// ---- CDNA5 async global->LDS path (guarded; sync fallback otherwise) ----
#if defined(__has_builtin)
#if __has_builtin(__builtin_amdgcn_global_load_async_to_lds_b128) && \
    __has_builtin(__builtin_amdgcn_s_wait_asynccnt)
#define USE_ASYNC_LDS 1
#endif
#endif

#ifdef USE_ASYNC_LDS
// builtin signature (from hipcc diagnostic): (int __vector(4) AS1*, int __vector(4) AS3*, Imm, Imm)
typedef int v4i_gcc __attribute__((vector_size(16)));
typedef __attribute__((address_space(1))) v4i_gcc g_v4i;
typedef __attribute__((address_space(3))) v4i_gcc l_v4i;
// copy 16 bytes global -> LDS per lane, tracked by ASYNCcnt
static __device__ __forceinline__ void async_cp16(const void* g, void* l) {
  __builtin_amdgcn_global_load_async_to_lds_b128((g_v4i*)g, (l_v4i*)l, 0, 0);
}
static __device__ __forceinline__ void async_wait0() {
  __builtin_amdgcn_s_wait_asynccnt(0);
}
#endif

// ---------- helpers ----------
static __device__ __forceinline__ unsigned short f2bf(float f) {
  unsigned int u = __builtin_bit_cast(unsigned int, f);
  u += 0x7FFFu + ((u >> 16) & 1u);  // round-to-nearest-even
  return (unsigned short)(u >> 16);
}

// build a v16bf WMMA operand from two contiguous 16-byte runs in LDS
static __device__ __forceinline__ v16bf frag16(const unsigned short* p0,
                                               const unsigned short* p1) {
  uint4 lo = *(const uint4*)p0;
  uint4 hi = *(const uint4*)p1;
  v8u u;
  u[0] = lo.x; u[1] = lo.y; u[2] = lo.z; u[3] = lo.w;
  u[4] = hi.x; u[5] = hi.y; u[6] = hi.z; u[7] = hi.w;
  return __builtin_bit_cast(v16bf, u);
}

// ---------- fp32 -> bf16 conversion ----------
__global__ __launch_bounds__(256) void f32_to_bf16_kernel(
    const float* __restrict__ src, unsigned short* __restrict__ dst, int n4) {
  int i = blockIdx.x * blockDim.x + threadIdx.x;
  if (i < n4) {
    float4 f = ((const float4*)src)[i];
    ushort4 o;
    o.x = f2bf(f.x); o.y = f2bf(f.y); o.z = f2bf(f.z); o.w = f2bf(f.w);
    ((ushort4*)dst)[i] = o;
  }
}

// ---------- GEMM: out[m,n] = sum_k A[m,k]*W[n,k] + bias[n] (+ocr[n]) ----------
// 256 threads = 8 waves. Block tile 128x64, wave tile 16x64 (4 WMMA accums),
// K step 32. Double-buffered async global->LDS when available.
template <bool ADD_OCR, bool OUT_BF16>
__global__ __launch_bounds__(256) void gemm_wmma_kernel(
    const unsigned short* __restrict__ A,   // [M,K] bf16
    const unsigned short* __restrict__ W,   // [N,K] bf16 (torch Linear weight)
    const float* __restrict__ bias,         // [N]
    const float* __restrict__ ocr,          // [N] or null
    unsigned short* __restrict__ outb,      // [M,N] bf16 (if OUT_BF16)
    float* __restrict__ outf,               // [M,N] f32  (else)
    int M, int N, int K) {
  __shared__ unsigned short As[2][128][32];  // 2 x 8 KB
  __shared__ unsigned short Bs[2][64][32];   // 2 x 4 KB

  const int tid = threadIdx.x;
  const int w = tid >> 5, lane = tid & 31;
  const int wm = w * 16;                 // wave's 16 M-rows within the 128 tile
  const int hi = lane >> 4, l15 = lane & 15;
  const int m0 = blockIdx.x * 128, n0 = blockIdx.y * 64;

  const int arow = tid >> 1, acol = (tid & 1) * 16;  // A: 128x32, 16 elem/thr
  const int brow = tid >> 2, bcol = (tid & 3) * 8;   // B: 64x32,   8 elem/thr

  const unsigned short* Ap = &A[(size_t)(m0 + arow) * K + acol];
  const unsigned short* Wp = &W[(size_t)(n0 + brow) * K + bcol];

  v8f acc[4] = {};

#ifdef USE_ASYNC_LDS
  // prologue: stage first K-tile
  async_cp16(Ap,     &As[0][arow][acol]);
  async_cp16(Ap + 8, &As[0][arow][acol + 8]);
  async_cp16(Wp,     &Bs[0][brow][bcol]);
  int buf = 0;
  for (int k0 = 0; k0 < K; k0 += 32, buf ^= 1) {
    async_wait0();
    __syncthreads();  // tile visible to all; all waves done reading buf^1
    if (k0 + 32 < K) {
      async_cp16(Ap + k0 + 32,     &As[buf ^ 1][arow][acol]);
      async_cp16(Ap + k0 + 32 + 8, &As[buf ^ 1][arow][acol + 8]);
      async_cp16(Wp + k0 + 32,     &Bs[buf ^ 1][brow][bcol]);
    }
    v16bf af = frag16(&As[buf][wm + l15][hi * 16], &As[buf][wm + l15][hi * 16 + 8]);
#pragma unroll
    for (int nb = 0; nb < 4; ++nb) {
      v16bf bf = frag16(&Bs[buf][nb * 16 + l15][hi * 16],
                        &Bs[buf][nb * 16 + l15][hi * 16 + 8]);
      acc[nb] = __builtin_amdgcn_wmma_f32_16x16x32_bf16(false, af, false, bf,
                                                        (short)0, acc[nb], false, false);
    }
  }
#else
  for (int k0 = 0; k0 < K; k0 += 32) {
    if (k0 + 32 < K) {  // prefetch next K-tile -> global_prefetch_b8
      __builtin_prefetch(Ap + k0 + 32, 0, 1);
      __builtin_prefetch(Wp + k0 + 32, 0, 1);
    }
    *(uint4*)&As[0][arow][acol]     = *(const uint4*)(Ap + k0);
    *(uint4*)&As[0][arow][acol + 8] = *(const uint4*)(Ap + k0 + 8);
    *(uint4*)&Bs[0][brow][bcol]     = *(const uint4*)(Wp + k0);
    __syncthreads();
    v16bf af = frag16(&As[0][wm + l15][hi * 16], &As[0][wm + l15][hi * 16 + 8]);
#pragma unroll
    for (int nb = 0; nb < 4; ++nb) {
      v16bf bf = frag16(&Bs[0][nb * 16 + l15][hi * 16],
                        &Bs[0][nb * 16 + l15][hi * 16 + 8]);
      acc[nb] = __builtin_amdgcn_wmma_f32_16x16x32_bf16(false, af, false, bf,
                                                        (short)0, acc[nb], false, false);
    }
    __syncthreads();
  }
#endif

#pragma unroll
  for (int nb = 0; nb < 4; ++nb) {
    const int n = n0 + nb * 16 + l15;
    const float badd = bias[n] + (ADD_OCR ? ocr[n] : 0.0f);
#pragma unroll
    for (int r = 0; r < 8; ++r) {
      const int m = m0 + wm + r + 8 * hi;  // C/D layout: vgpr r, half hi
      const float v = acc[nb][r] + badd;
      if (OUT_BF16) outb[(size_t)m * N + n] = f2bf(v);
      else          outf[(size_t)m * N + n] = v;
    }
  }
}

// ---------- flash attention with band mask ----------
// Block: 256 thr (8 waves), one (b,h), 32 query rows. Key tiles of 64.
// wq = wave&1 (query 16-row half), wn = wave>>1 (key/dh 16-col quadrant).
__global__ __launch_bounds__(256) void flash_attn_kernel(
    const unsigned short* __restrict__ Q,
    const unsigned short* __restrict__ Km,
    const unsigned short* __restrict__ Vm,
    unsigned short* __restrict__ O) {
  __shared__ unsigned short Qs[32][64];
  __shared__ unsigned short Ks[64][64];
  __shared__ unsigned short Vts[64][64];  // transposed: [d][key]
  __shared__ unsigned short Ps[32][64];
  __shared__ float redmax[32][4];
  __shared__ float redsum[32][4];

  const int tid = threadIdx.x;
  const int w = tid >> 5, lane = tid & 31;
  const int wq = w & 1, wn = w >> 1;  // wn 0..3
  const int hi = lane >> 4, l15 = lane & 15;
  const int b = blockIdx.z, h = blockIdx.y;
  const int q0 = blockIdx.x * 32;
  const int headoff = h * HEAD_DIM;

  {  // load Q tile 32x64
    const int row = tid >> 3, d0 = (tid & 7) * 8;
    const unsigned short* qp =
        &Q[((size_t)(b * SEQ + q0 + row)) * D_MODEL + headoff + d0];
#ifdef USE_ASYNC_LDS
    async_cp16(qp, &Qs[row][d0]);
    async_wait0();
#else
    *(uint4*)&Qs[row][d0] = *(const uint4*)qp;
#endif
  }
  __syncthreads();

  // Q fragments for the two 32-wide dh chunks, reused across all key tiles
  v16bf qa0 = frag16(&Qs[wq * 16 + l15][hi * 16],      &Qs[wq * 16 + l15][hi * 16 + 8]);
  v16bf qa1 = frag16(&Qs[wq * 16 + l15][32 + hi * 16], &Qs[wq * 16 + l15][32 + hi * 16 + 8]);

  v8f oacc = {};
  float mrow[8], lrow[8];
#pragma unroll
  for (int r = 0; r < 8; ++r) { mrow[r] = -1e30f; lrow[r] = 0.0f; }

  const int ldRow = tid >> 2, ldCol = (tid & 3) * 16;

  for (int t = 0; t < SEQ / 64; ++t) {
    const int k0 = t * 64;
    {  // stage K (row-major, async if available) and V (transposed via VGPRs)
      const unsigned short* kp =
          &Km[((size_t)(b * SEQ + k0 + ldRow)) * D_MODEL + headoff + ldCol];
#ifdef USE_ASYNC_LDS
      async_cp16(kp,     &Ks[ldRow][ldCol]);
      async_cp16(kp + 8, &Ks[ldRow][ldCol + 8]);
#else
      *(uint4*)&Ks[ldRow][ldCol]     = *(const uint4*)kp;
      *(uint4*)&Ks[ldRow][ldCol + 8] = *(const uint4*)(kp + 8);
#endif
      const unsigned short* vp =
          &Vm[((size_t)(b * SEQ + k0 + ldRow)) * D_MODEL + headoff + ldCol];
      u16x8 va = __builtin_bit_cast(u16x8, *(const uint4*)vp);
      u16x8 vb = __builtin_bit_cast(u16x8, *(const uint4*)(vp + 8));
#pragma unroll
      for (int j = 0; j < 8; ++j) {
        Vts[ldCol + j][ldRow]     = va[j];
        Vts[ldCol + 8 + j][ldRow] = vb[j];
      }
#ifdef USE_ASYNC_LDS
      async_wait0();
#endif
    }
    __syncthreads();

    // S = Q K^T : B operand column n = key, K-dim = dh (contiguous in Ks rows)
    v8f sacc = {};
    {
      v16bf bk0 = frag16(&Ks[wn * 16 + l15][hi * 16],      &Ks[wn * 16 + l15][hi * 16 + 8]);
      sacc = __builtin_amdgcn_wmma_f32_16x16x32_bf16(false, qa0, false, bk0, (short)0, sacc, false, false);
      v16bf bk1 = frag16(&Ks[wn * 16 + l15][32 + hi * 16], &Ks[wn * 16 + l15][32 + hi * 16 + 8]);
      sacc = __builtin_amdgcn_wmma_f32_16x16x32_bf16(false, qa1, false, bk1, (short)0, sacc, false, false);
    }

    // scale + band mask, per-wave row max over the 16 lanes owning a row
    float sv[8], tmax[8];
    const int kcol = k0 + wn * 16 + l15;
#pragma unroll
    for (int r = 0; r < 8; ++r) {
      const int qrow = q0 + wq * 16 + r + 8 * hi;
      int dd = qrow - kcol; if (dd < 0) dd = -dd;
      float v = sacc[r] * 0.125f + (dd <= 2 ? 0.1f : 0.0f);
      sv[r] = v;
      v = fmaxf(v, __shfl_xor(v, 1));
      v = fmaxf(v, __shfl_xor(v, 2));
      v = fmaxf(v, __shfl_xor(v, 4));
      v = fmaxf(v, __shfl_xor(v, 8));
      tmax[r] = v;
    }
    if (l15 == 0) {
#pragma unroll
      for (int r = 0; r < 8; ++r) redmax[wq * 16 + r + 8 * hi][wn] = tmax[r];
    }
    __syncthreads();

    float newm[8], psum[8];
#pragma unroll
    for (int r = 0; r < 8; ++r) {
      const int row = wq * 16 + r + 8 * hi;
      float tm = fmaxf(fmaxf(redmax[row][0], redmax[row][1]),
                       fmaxf(redmax[row][2], redmax[row][3]));
      newm[r] = fmaxf(mrow[r], tm);
      float p = __expf(sv[r] - newm[r]);
      Ps[row][wn * 16 + l15] = f2bf(p);
      float s = p;
      s += __shfl_xor(s, 1);
      s += __shfl_xor(s, 2);
      s += __shfl_xor(s, 4);
      s += __shfl_xor(s, 8);
      psum[r] = s;
    }
    if (l15 == 0) {
#pragma unroll
      for (int r = 0; r < 8; ++r) redsum[wq * 16 + r + 8 * hi][wn] = psum[r];
    }
    __syncthreads();

#pragma unroll
    for (int r = 0; r < 8; ++r) {
      const int row = wq * 16 + r + 8 * hi;
      const float ts = redsum[row][0] + redsum[row][1] + redsum[row][2] + redsum[row][3];
      const float alpha = __expf(mrow[r] - newm[r]);
      lrow[r] = lrow[r] * alpha + ts;
      oacc[r] *= alpha;
      mrow[r] = newm[r];
    }

    // O += P V : A = P (rows wq*16..), B = V^T staged (n = dh col, K = key)
    {
      v16bf pa0 = frag16(&Ps[wq * 16 + l15][hi * 16],      &Ps[wq * 16 + l15][hi * 16 + 8]);
      v16bf bv0 = frag16(&Vts[wn * 16 + l15][hi * 16],    &Vts[wn * 16 + l15][hi * 16 + 8]);
      oacc = __builtin_amdgcn_wmma_f32_16x16x32_bf16(false, pa0, false, bv0, (short)0, oacc, false, false);
      v16bf pa1 = frag16(&Ps[wq * 16 + l15][32 + hi * 16], &Ps[wq * 16 + l15][32 + hi * 16 + 8]);
      v16bf bv1 = frag16(&Vts[wn * 16 + l15][32 + hi * 16], &Vts[wn * 16 + l15][32 + hi * 16 + 8]);
      oacc = __builtin_amdgcn_wmma_f32_16x16x32_bf16(false, pa1, false, bv1, (short)0, oacc, false, false);
    }
    __syncthreads();
  }

#pragma unroll
  for (int r = 0; r < 8; ++r) {
    const int m = q0 + wq * 16 + r + 8 * hi;
    const int col = headoff + wn * 16 + l15;
    O[((size_t)(b * SEQ + m)) * D_MODEL + col] = f2bf(oacc[r] / lrow[r]);
  }
}

// ---------- host ----------
extern "C" void kernel_launch(void* const* d_in, const int* in_sizes, int n_in,
                              void* d_out, int out_size, void* d_ws, size_t ws_size,
                              hipStream_t stream) {
  (void)in_sizes; (void)n_in; (void)out_size; (void)ws_size;
  const float* x   = (const float*)d_in[0];
  const float* Wq  = (const float*)d_in[2];
  const float* bq  = (const float*)d_in[3];
  const float* Wk  = (const float*)d_in[4];
  const float* bk  = (const float*)d_in[5];
  const float* Wv  = (const float*)d_in[6];
  const float* bv  = (const float*)d_in[7];
  const float* Wo  = (const float*)d_in[8];
  const float* bo  = (const float*)d_in[9];
  const float* ocr = (const float*)d_in[10];
  float* out = (float*)d_out;

  const size_t NX = (size_t)MROWS * D_MODEL;    // 8Mi elems
  const size_t NW = (size_t)D_MODEL * D_MODEL;  // 1Mi elems
  char* p = (char*)d_ws;
  unsigned short* xb  = (unsigned short*)p; p += NX * 2;
  unsigned short* Wqb = (unsigned short*)p; p += NW * 2;
  unsigned short* Wkb = (unsigned short*)p; p += NW * 2;
  unsigned short* Wvb = (unsigned short*)p; p += NW * 2;
  unsigned short* Wob = (unsigned short*)p; p += NW * 2;
  unsigned short* Qb  = (unsigned short*)p; p += NX * 2;
  unsigned short* Kb  = (unsigned short*)p; p += NX * 2;
  unsigned short* Vb  = (unsigned short*)p; p += NX * 2;
  unsigned short* Ab  = (unsigned short*)p; p += NX * 2;  // total ~88 MB

  // 1) fp32 -> bf16
  f32_to_bf16_kernel<<<(int)(NX / 4 + 255) / 256, 256, 0, stream>>>(x,  xb,  (int)(NX / 4));
  f32_to_bf16_kernel<<<(int)(NW / 4 + 255) / 256, 256, 0, stream>>>(Wq, Wqb, (int)(NW / 4));
  f32_to_bf16_kernel<<<(int)(NW / 4 + 255) / 256, 256, 0, stream>>>(Wk, Wkb, (int)(NW / 4));
  f32_to_bf16_kernel<<<(int)(NW / 4 + 255) / 256, 256, 0, stream>>>(Wv, Wvb, (int)(NW / 4));
  f32_to_bf16_kernel<<<(int)(NW / 4 + 255) / 256, 256, 0, stream>>>(Wo, Wob, (int)(NW / 4));

  // 2) Q/K/V projections (bias fused; ocr_bias added to Q)
  dim3 ggrid(MROWS / 128, D_MODEL / 64);
  gemm_wmma_kernel<true,  true><<<ggrid, 256, 0, stream>>>(xb, Wqb, bq, ocr, Qb, nullptr, MROWS, D_MODEL, D_MODEL);
  gemm_wmma_kernel<false, true><<<ggrid, 256, 0, stream>>>(xb, Wkb, bk, nullptr, Kb, nullptr, MROWS, D_MODEL, D_MODEL);
  gemm_wmma_kernel<false, true><<<ggrid, 256, 0, stream>>>(xb, Wvb, bv, nullptr, Vb, nullptr, MROWS, D_MODEL, D_MODEL);

  // 3) banded-softmax flash attention
  flash_attn_kernel<<<dim3(SEQ / 32, N_HEADS, BATCH), 256, 0, stream>>>(Qb, Kb, Vb, Ab);

  // 4) output projection -> fp32 d_out
  gemm_wmma_kernel<false, false><<<ggrid, 256, 0, stream>>>(Ab, Wob, bo, nullptr, nullptr, out, MROWS, D_MODEL, D_MODEL);
}
